// BiLSTM_CRF_16045997817976
// MI455X (gfx1250) — matrix operand order repaired and checked by reading, not compiled, not run on hardware
//
#include <hip/hip_runtime.h>
#include <hip/hip_bf16.h>

// ---------------------------------------------------------------------------
// BiLSTM-CRF for MI455X (gfx1250, wave32, WMMA).
// Heavy math: v_wmma_f32_16x16x32_bf16.  Whh/Wih converted to bf16 (24MB total,
// L2-resident).  The 256 recurrent steps run inside ONE persistent kernel with
// a per-direction device-scope atomic barrier; the K loop is software-
// pipelined (next chunk's loads issued before current chunk's WMMAs).
// ---------------------------------------------------------------------------

typedef __bf16 bf16_t;
typedef __attribute__((ext_vector_type(16))) __bf16 v16bf;
typedef __attribute__((ext_vector_type(8)))  float  v8f;

#define TT 256     // seq len
#define BB 32      // batch
#define HH 1024    // hidden per direction
#define DD 330     // DW + CO
#define DP 352     // DD padded to multiple of 32 (11*32)
#define DWC 300    // word dim
#define COC 30     // char-CNN out channels
#define LLC 20     // chars per word
#define DCC 30     // char emb dim
#define KK  19     // tagset
#define TSTART 17
#define TSTOP  18
#define NBLK 16    // h-col blocks per direction (64 cols each)

__device__ __forceinline__ float bf2f(bf16_t h) { return (float)h; }
__device__ __forceinline__ bf16_t f2bf(float f) { return (bf16_t)f; }

// A operand: lane<16 holds K 0..15, lane>=16 holds K 16..31
__device__ __forceinline__ v16bf load_a(const bf16_t* base, int lhalf) {
    union { v16bf v; uint4 q[2]; } u;
    u.q[0] = *reinterpret_cast<const uint4*>(base + lhalf);
    u.q[1] = *reinterpret_cast<const uint4*>(base + lhalf + 8);
    return u.v;
}
// B operand: lane<16 holds K 0..15, lane>=16 holds K 16..31 (contiguous 32B)
__device__ __forceinline__ v16bf load_b(const bf16_t* base, int bhalf) {
    union { v16bf v; uint4 q[2]; } u;
    const uint4* p = reinterpret_cast<const uint4*>(base + bhalf);
    u.q[0] = p[0];
    u.q[1] = p[1];
    return u.v;
}

__device__ __forceinline__ v8f wmma_bf16(v16bf a, v16bf b, v8f c) {
    return __builtin_amdgcn_wmma_f32_16x16x32_bf16(false, a, false, b,
                                                   (short)0, c, false, false);
}

__device__ __forceinline__ float sigm(float x) { return 1.f / (1.f + __expf(-x)); }

// Software-pipelined 4-gate GEMM segment: 16x16 tile per gate, K = kmax.
// Next chunk's A/B loads are issued BEFORE current chunk's WMMAs so VMEM
// overlaps the matrix pipe instead of s_wait_loadcnt 0 before every WMMA.
__device__ __forceinline__ void gemm_seg(const bf16_t* __restrict__ arow,
                                         const bf16_t* __restrict__ b0r,
                                         const bf16_t* __restrict__ b1r,
                                         const bf16_t* __restrict__ b2r,
                                         const bf16_t* __restrict__ b3r,
                                         int kmax, int lhalf, int bhalf,
                                         v8f acc[4]) {
    v16bf a  = load_a(arow, lhalf);
    v16bf b0 = load_b(b0r, bhalf);
    v16bf b1 = load_b(b1r, bhalf);
    v16bf b2 = load_b(b2r, bhalf);
    v16bf b3 = load_b(b3r, bhalf);
    #pragma unroll 2
    for (int k0 = 32; k0 < kmax; k0 += 32) {
        v16bf an  = load_a(arow + k0, lhalf);
        v16bf bn0 = load_b(b0r + k0, bhalf);
        v16bf bn1 = load_b(b1r + k0, bhalf);
        v16bf bn2 = load_b(b2r + k0, bhalf);
        v16bf bn3 = load_b(b3r + k0, bhalf);
        acc[0] = wmma_bf16(a, b0, acc[0]);
        acc[1] = wmma_bf16(a, b1, acc[1]);
        acc[2] = wmma_bf16(a, b2, acc[2]);
        acc[3] = wmma_bf16(a, b3, acc[3]);
        a = an; b0 = bn0; b1 = bn1; b2 = bn2; b3 = bn3;
    }
    acc[0] = wmma_bf16(a, b0, acc[0]);
    acc[1] = wmma_bf16(a, b1, acc[1]);
    acc[2] = wmma_bf16(a, b2, acc[2]);
    acc[3] = wmma_bf16(a, b3, acc[3]);
}

// ------------------------- weight conversion -------------------------------
__global__ void convert_wih_kernel(const float* __restrict__ Wf,
                                   const float* __restrict__ Wb,
                                   bf16_t* __restrict__ out) {  // [2][4H][DP]
    size_t i = (size_t)blockIdx.x * blockDim.x + threadIdx.x;
    size_t total = (size_t)2 * 4 * HH * DP;
    if (i >= total) return;
    int    col = (int)(i % DP);
    size_t row = (i / DP) % (4 * HH);
    int    dir = (int)(i / ((size_t)4 * HH * DP));
    const float* W = dir ? Wb : Wf;
    out[i] = (col < DD) ? f2bf(W[row * DD + col]) : f2bf(0.f);
}

__global__ void convert_whh_kernel(const float* __restrict__ Wf,
                                   const float* __restrict__ Wb,
                                   bf16_t* __restrict__ out) {  // [2][4H][H]
    size_t i = (size_t)blockIdx.x * blockDim.x + threadIdx.x;
    size_t per = (size_t)4 * HH * HH;
    if (i >= 2 * per) return;
    out[i] = f2bf(i < per ? Wf[i] : Wb[i - per]);
}

__global__ void make_bias_kernel(const float* bih_f, const float* bhh_f,
                                 const float* bih_b, const float* bhh_b,
                                 float* __restrict__ out) {     // [2][4H]
    int i = blockIdx.x * blockDim.x + threadIdx.x;
    if (i >= 2 * 4 * HH) return;
    int j = i % (4 * HH);
    out[i] = (i < 4 * HH) ? (bih_f[j] + bhh_f[j]) : (bih_b[j] + bhh_b[j]);
}

__global__ void init_barrier_kernel(unsigned* bar) {
    if (threadIdx.x < 2) bar[(size_t)threadIdx.x * 64] = 0u;
}

// --------------------- embedding + char CNN --> x (bf16) -------------------
__global__ void __launch_bounds__(128)
embed_cnn_kernel(const int* __restrict__ words,      // [B][T]
                 const int* __restrict__ chars,      // [B][T][L]
                 const float* __restrict__ word_emb, // [V][DW]
                 const float* __restrict__ char_emb, // [CV][DC]
                 const float* __restrict__ conv_w,   // [CO][3][DC]
                 const float* __restrict__ conv_b,   // [CO]
                 bf16_t* __restrict__ x) {           // [T][B][DP]
    const int bt = blockIdx.x;          // b*T + t
    const int b = bt / TT, t = bt % TT;
    const int tid = threadIdx.x;
    __shared__ float ce[LLC][DCC];
    for (int i = tid; i < LLC * DCC; i += 128) {
        int l = i / DCC, d = i % DCC;
        ce[l][d] = char_emb[(size_t)chars[(size_t)bt * LLC + l] * DCC + d];
    }
    bf16_t* xr = x + ((size_t)t * BB + b) * DP;
    const int widx = words[bt];
    for (int d = tid; d < DP; d += 128) {
        if (d < DWC)            xr[d] = f2bf(word_emb[(size_t)widx * DWC + d]);
        else if (d >= DWC + COC) xr[d] = f2bf(0.f);
    }
    __syncthreads();
    if (tid < COC) {
        float best = -1e30f;
        for (int p = 0; p < LLC + 2; ++p) {          // 22 output positions
            float s = 0.f;
            #pragma unroll
            for (int kh = 0; kh < 3; ++kh) {
                int l = p - 2 + kh;
                if (l >= 0 && l < LLC) {
                    const float* wv = conv_w + ((size_t)tid * 3 + kh) * DCC;
                    for (int d = 0; d < DCC; ++d) s += ce[l][d] * wv[d];
                }
            }
            best = fmaxf(best, s);
        }
        xr[DWC + tid] = f2bf(best + conv_b[tid]);
    }
}

// --------------------- persistent BiLSTM (all 256 steps) -------------------
// grid (NBLK, 2): 16 blocks of 64 h-cols per direction; 256 thr = 8 waves.
// wave w: row-tile m0 = (w&1)*16, col-tile n0 = blk*64 + (w>>1)*16; each wave
// holds all 4 gate tiles so the epilogue is shuffle-free.  Blocks of one
// direction sync each step on a device-scope atomic counter.
__global__ void __launch_bounds__(256)
lstm_persistent_kernel(const bf16_t* __restrict__ x,    // [T][B][DP]
                       const bf16_t* __restrict__ Wih,  // [2][4H][DP]
                       const bf16_t* __restrict__ Whh,  // [2][4H][H]
                       const float*  __restrict__ bias, // [2][4H]
                       bf16_t*       h_all,             // [2][T][B][H]
                       float*        c_state,           // [2][B][H]
                       unsigned*     bar_all) {         // [2] stride-64
    const int dir  = blockIdx.y;
    unsigned* bar  = bar_all + (size_t)dir * 64;
    const int wave = threadIdx.x >> 5;
    const int lane = threadIdx.x & 31;
    const int m0   = (wave & 1) * 16;
    const int n0   = blockIdx.x * 64 + (wave >> 1) * 16;
    const int lm   = lane & 15;
    const int lhalf = (lane & 16) ? 16 : 0;   // A-operand half
    const int bhalf = (lane & 16) ? 16 : 0;   // B-operand half

    const bf16_t* Wih_d  = Wih  + (size_t)dir * 4 * HH * DP;
    const bf16_t* Whh_d  = Whh  + (size_t)dir * 4 * HH * HH;
    const float*  bias_d = bias + (size_t)dir * 4 * HH;

    // loop-invariant operand bases (gate offsets fold into IOFFSET imm)
    const bf16_t* wx = Wih_d + (size_t)(n0 + lm) * DP;
    const bf16_t* wh = Whh_d + (size_t)(n0 + lm) * HH;

    const int mb = m0 + ((lane & 16) ? 8 : 0);
    const int nn = n0 + lm;
    const float bi = bias_d[0 * HH + nn];
    const float bf = bias_d[1 * HH + nn];
    const float bg = bias_d[2 * HH + nn];
    const float bo = bias_d[3 * HH + nn];
    float* cptr = c_state + (size_t)dir * BB * HH;

    for (int step = 0; step < TT; ++step) {
        const int t = dir ? (TT - 1 - step) : step;
        v8f acc[4] = {v8f{}, v8f{}, v8f{}, v8f{}};

        // input projection: K = DP
        gemm_seg(x + ((size_t)t * BB + (m0 + lm)) * DP,
                 wx, wx + (size_t)1 * HH * DP, wx + (size_t)2 * HH * DP,
                 wx + (size_t)3 * HH * DP, DP, lhalf, bhalf, acc);
        // recurrent: K = H (skipped at first step: h = 0)
        if (step > 0) {
            const int tprev = dir ? (t + 1) : (t - 1);
            gemm_seg(h_all + (((size_t)dir * TT + tprev) * BB + (m0 + lm)) * HH,
                     wh, wh + (size_t)1 * HH * HH, wh + (size_t)2 * HH * HH,
                     wh + (size_t)3 * HH * HH, HH, lhalf, bhalf, acc);
        }

        // gate epilogue: each lane owns 8 unique (b, hcol) cells
        bf16_t* hout = h_all + ((size_t)dir * TT + t) * BB * HH;
        #pragma unroll
        for (int r = 0; r < 8; ++r) {
            const int m = mb + r;
            float gi = acc[0][r] + bi;
            float gf = acc[1][r] + bf;
            float gg = acc[2][r] + bg;
            float go = acc[3][r] + bo;
            float cprev = (step > 0) ? cptr[(size_t)m * HH + nn] : 0.f;
            float c = sigm(gf) * cprev + sigm(gi) * tanhf(gg);
            cptr[(size_t)m * HH + nn] = c;
            hout[(size_t)m * HH + nn] = f2bf(sigm(go) * tanhf(c));
        }

        // direction-wide barrier: release our h stores, wait for all NBLK
        __threadfence();
        __syncthreads();
        if (threadIdx.x == 0) {
            __hip_atomic_fetch_add(bar, 1u, __ATOMIC_RELEASE,
                                   __HIP_MEMORY_SCOPE_AGENT);
            const unsigned target = (unsigned)NBLK * (unsigned)(step + 1);
            while (__hip_atomic_load(bar, __ATOMIC_ACQUIRE,
                                     __HIP_MEMORY_SCOPE_AGENT) < target)
                __builtin_amdgcn_s_sleep(1);
        }
        __syncthreads();
    }
}

// ----------------------------- emit projection -----------------------------
__global__ void emit_kernel(const bf16_t* __restrict__ h_all,  // [2][T][B][H]
                            const float* __restrict__ W_out,   // [K][2H]
                            const float* __restrict__ b_out,   // [K]
                            float* __restrict__ emit) {        // [T][B][K]
    const int idx = blockIdx.x * blockDim.x + threadIdx.x;
    if (idx >= TT * BB * KK) return;
    const int k  = idx % KK;
    const int tb = idx / KK;    // t*B + b
    const bf16_t* hf = h_all + (size_t)tb * HH;
    const bf16_t* hb = h_all + ((size_t)TT * BB + tb) * HH;
    const float* w = W_out + (size_t)k * 2 * HH;
    float s = b_out[k];
    for (int j = 0; j < HH; ++j)
        s += bf2f(hf[j]) * w[j] + bf2f(hb[j]) * w[HH + j];
    emit[idx] = s;
}

// --------------------------- CRF forward + gold ----------------------------
__global__ void __launch_bounds__(608)
crf_kernel(const float* __restrict__ emit,   // [T][B][K]
           const float* __restrict__ trans,  // [K][K]  trans[next][prev]
           const int*   __restrict__ tags,   // [B][T]
           float* __restrict__ out) {
    __shared__ float alpha[BB][KK];
    __shared__ float ts[KK][KK];
    __shared__ float red[BB];
    const int tid = threadIdx.x;
    for (int i = tid; i < KK * KK; i += blockDim.x) ts[i / KK][i % KK] = trans[i];
    const int b  = tid & 31;
    const int kn = tid >> 5;          // 0..18 (608 = 19 waves)
    alpha[b][kn] = (kn == TSTART) ? 0.f : -10000.f;
    __syncthreads();
    for (int t = 0; t < TT; ++t) {
        float mx = -1e30f;
        #pragma unroll
        for (int kp = 0; kp < KK; ++kp)
            mx = fmaxf(mx, alpha[b][kp] + ts[kn][kp]);
        float s = 0.f;
        #pragma unroll
        for (int kp = 0; kp < KK; ++kp)
            s += __expf(alpha[b][kp] + ts[kn][kp] - mx);
        float anew = mx + __logf(s) + emit[((size_t)t * BB + b) * KK + kn];
        __syncthreads();
        alpha[b][kn] = anew;
        __syncthreads();
    }
    if (kn == 0) {
        float mx = -1e30f;
        for (int kp = 0; kp < KK; ++kp)
            mx = fmaxf(mx, alpha[b][kp] + ts[TSTOP][kp]);
        float s = 0.f;
        for (int kp = 0; kp < KK; ++kp)
            s += __expf(alpha[b][kp] + ts[TSTOP][kp] - mx);
        const float logZ = mx + __logf(s);
        float esc = 0.f, tsc = 0.f;
        int prev = TSTART;
        for (int t = 0; t < TT; ++t) {
            int tg = tags[(size_t)b * TT + t];
            esc += emit[((size_t)t * BB + b) * KK + tg];
            tsc += ts[tg][prev];
            prev = tg;
        }
        tsc += ts[TSTOP][prev];
        red[b] = logZ - (esc + tsc);
    }
    __syncthreads();
    if (tid == 0) {
        float s = 0.f;
        for (int i = 0; i < BB; ++i) s += red[i];
        out[0] = s;
    }
}

// ------------------------------- launcher ----------------------------------
extern "C" void kernel_launch(void* const* d_in, const int* in_sizes, int n_in,
                              void* d_out, int out_size, void* d_ws, size_t ws_size,
                              hipStream_t stream) {
    (void)in_sizes; (void)n_in; (void)out_size; (void)ws_size;
    const int*   words    = (const int*)d_in[0];
    const int*   chars    = (const int*)d_in[1];
    const int*   tags     = (const int*)d_in[2];
    const float* word_emb = (const float*)d_in[3];
    const float* char_emb = (const float*)d_in[4];
    const float* conv_w   = (const float*)d_in[5];
    const float* conv_b   = (const float*)d_in[6];
    const float* Wih_f    = (const float*)d_in[7];
    const float* Whh_f    = (const float*)d_in[8];
    const float* bih_f    = (const float*)d_in[9];
    const float* bhh_f    = (const float*)d_in[10];
    const float* Wih_b    = (const float*)d_in[11];
    const float* Whh_b    = (const float*)d_in[12];
    const float* bih_b    = (const float*)d_in[13];
    const float* bhh_b    = (const float*)d_in[14];
    const float* W_out    = (const float*)d_in[15];
    const float* b_out    = (const float*)d_in[16];
    const float* trans    = (const float*)d_in[17];

    char* ws = (char*)d_ws;
    auto au = [](size_t x) { return (x + 255) & ~(size_t)255; };
    size_t off = 0;
    bf16_t*   x_bf    = (bf16_t*)(ws + off); off += au((size_t)TT * BB * DP * 2);
    bf16_t*   Wih_bf  = (bf16_t*)(ws + off); off += au((size_t)2 * 4 * HH * DP * 2);
    bf16_t*   Whh_bf  = (bf16_t*)(ws + off); off += au((size_t)2 * 4 * HH * HH * 2);
    float*    bias    = (float*) (ws + off); off += au((size_t)2 * 4 * HH * 4);
    bf16_t*   h_all   = (bf16_t*)(ws + off); off += au((size_t)2 * TT * BB * HH * 2);
    float*    c_state = (float*) (ws + off); off += au((size_t)2 * BB * HH * 4);
    float*    emitb   = (float*) (ws + off); off += au((size_t)TT * BB * KK * 4);
    unsigned* barrier = (unsigned*)(ws + off); off += au((size_t)2 * 64 * 4);

    {   size_t total = (size_t)2 * 4 * HH * DP;
        convert_wih_kernel<<<(unsigned)((total + 255) / 256), 256, 0, stream>>>(
            Wih_f, Wih_b, Wih_bf); }
    {   size_t total = (size_t)2 * 4 * HH * HH;
        convert_whh_kernel<<<(unsigned)((total + 255) / 256), 256, 0, stream>>>(
            Whh_f, Whh_b, Whh_bf); }
    make_bias_kernel<<<(2 * 4 * HH + 255) / 256, 256, 0, stream>>>(
        bih_f, bhh_f, bih_b, bhh_b, bias);
    init_barrier_kernel<<<1, 64, 0, stream>>>(barrier);
    embed_cnn_kernel<<<TT * BB, 128, 0, stream>>>(
        words, chars, word_emb, char_emb, conv_w, conv_b, x_bf);

    lstm_persistent_kernel<<<dim3(NBLK, 2), 256, 0, stream>>>(
        x_bf, Wih_bf, Whh_bf, bias, h_all, c_state, barrier);

    emit_kernel<<<(TT * BB * KK + 255) / 256, 256, 0, stream>>>(
        h_all, W_out, b_out, emitb);
    crf_kernel<<<1, 608, 0, stream>>>(emitb, trans, tags, (float*)d_out);
}
